// ConvalueTransformer_55886114456053
// MI455X (gfx1250) — matrix operand order repaired and checked by reading, not compiled, not run on hardware
//
#include <hip/hip_runtime.h>
#include <hip/hip_bf16.h>

// ---------------------------------------------------------------------------
// ConvalueTransformer on MI455X (gfx1250), wave32, bf16 WMMA everywhere.
// Round 3: fix async global->LDS builtin argument types (int4 pointers with
// explicit address-space qualifiers, matching the builtin's *1/*3 signature).
// ---------------------------------------------------------------------------

#define Bn 16
#define Nn 512
#define DIMn 64
#define HEADSn 8
#define DHn 64
#define FFn 256
#define CDIMn 124
#define VDIMn 992
#define ROWS (Bn * Nn)          // 8192
#define QKW (2 * HEADSn * DHn)  // 1024

typedef __attribute__((ext_vector_type(16))) __bf16 v16bf;
typedef __attribute__((ext_vector_type(8)))  float  v8f;

union FragU {
    v16bf v;
    unsigned short s[16];
    unsigned int   u[8];
};

struct U128 { unsigned int x[4]; };

#if __has_builtin(__builtin_amdgcn_global_load_async_to_lds_b128) && \
    __has_builtin(__builtin_amdgcn_s_wait_asynccnt)
#define HAVE_ASYNC_LDS 1
typedef int v4i_t __attribute__((ext_vector_type(4)));
typedef v4i_t __attribute__((address_space(1)))* gv4i_p;   // global int4*
typedef v4i_t __attribute__((address_space(3)))* lv4i_p;   // LDS int4*
#endif

__device__ __forceinline__ unsigned short f2b(float f) {
    union { float f; unsigned u; } c; c.f = f;
    unsigned u = c.u;
    u += 0x7FFFu + ((u >> 16) & 1u);   // round to nearest even
    return (unsigned short)(u >> 16);
}
__device__ __forceinline__ float b2f(unsigned short s) {
    union { unsigned u; float f; } c; c.u = ((unsigned)s) << 16;
    return c.f;
}

// ---------------------------------------------------------------------------
// f32 -> bf16 transpose conversion for weights: in [depth,K,N] -> out [depth,N,K]
// ---------------------------------------------------------------------------
__global__ void cvt_transpose_kernel(const float* __restrict__ in,
                                     unsigned short* __restrict__ out,
                                     int K, int N, int total) {
    int idx = blockIdx.x * blockDim.x + threadIdx.x;
    if (idx >= total) return;
    int n = idx % N;
    int k = (idx / N) % K;
    int l = idx / (N * K);
    out[((size_t)l * N + n) * K + k] = f2b(in[idx]);
}

// ---------------------------------------------------------------------------
// LayerNorm over DIM=64: one wave32 per row, 2 elems per lane. Output bf16.
// ---------------------------------------------------------------------------
__global__ void ln_kernel(const float* __restrict__ x,
                          const float* __restrict__ g,
                          const float* __restrict__ bt,
                          unsigned short* __restrict__ out, int rows) {
    int warp = (blockIdx.x * blockDim.x + threadIdx.x) >> 5;
    int lane = threadIdx.x & 31;
    if (warp >= rows) return;
    const float* xr = x + (size_t)warp * DIMn;
    float a = xr[lane], c = xr[lane + 32];
    float s = a + c;
    for (int m = 16; m >= 1; m >>= 1) s += __shfl_xor(s, m, 32);
    float mean = s * (1.0f / 64.0f);
    float da = a - mean, dc = c - mean;
    float vv = da * da + dc * dc;
    for (int m = 16; m >= 1; m >>= 1) vv += __shfl_xor(vv, m, 32);
    float inv = rsqrtf(vv * (1.0f / 64.0f) + 1e-5f);
    out[(size_t)warp * DIMn + lane]      = f2b(da * inv * g[lane] + bt[lane]);
    out[(size_t)warp * DIMn + lane + 32] = f2b(dc * inv * g[lane + 32] + bt[lane + 32]);
}

// ---------------------------------------------------------------------------
// Generic bf16 GEMM via WMMA. A:[M,K] bf16 row-major, Bt:[N,K] bf16 (B^T).
// The whole weight matrix is staged into LDS once per block (async global->LDS
// on gfx1250 when available), then all frag reads are contiguous b128s.
// EPI 0: store bf16. EPI 1: f32 out = acc + bias[col] + residual (in place).
// EPI 2: bf16 out = gelu(acc + bias[col])   (exact gelu, erf).
// ---------------------------------------------------------------------------
template <int EPI>
__global__ void gemm_wmma(const unsigned short* __restrict__ A,
                          const unsigned short* __restrict__ Bt,
                          int M, int N, int K,
                          const float* __restrict__ bias,
                          const float* __restrict__ res,
                          float* __restrict__ outF,
                          unsigned short* __restrict__ outH) {
    extern __shared__ unsigned short Blds[];        // N*K bf16
    // ---- stage B^T into LDS (linear 16B-chunk copy by all threads) ------
    {
        int nchunks = (N * K) >> 3;                 // 16-byte chunks
        for (int c = threadIdx.x; c < nchunks; c += blockDim.x) {
#ifdef HAVE_ASYNC_LDS
            __builtin_amdgcn_global_load_async_to_lds_b128(
                (gv4i_p)(Bt + (size_t)c * 8),
                (lv4i_p)(Blds + (size_t)c * 8), 0, 0);
#else
            ((U128*)Blds)[c] = ((const U128*)Bt)[c];
#endif
        }
#ifdef HAVE_ASYNC_LDS
        __builtin_amdgcn_s_wait_asynccnt(0);
#endif
        __syncthreads();
    }

    int wave = (blockIdx.x * blockDim.x + threadIdx.x) >> 5;
    int lane = threadIdx.x & 31;
    int tilesN = N >> 4;
    int tiles = (M >> 4) * tilesN;
    if (wave >= tiles) return;
    int tm = wave / tilesN, tn = wave % tilesN;
    int lr = lane & 15, grp = lane >> 4;

    int arow = (tm << 4) + lr;                 // A-frag: row = lane&15
    int col  = (tn << 4) + lr;                 // B/D-frag: col = lane&15
    const unsigned int* ap =
        reinterpret_cast<const unsigned int*>(A + (size_t)arow * K);
    const unsigned int* bp =
        reinterpret_cast<const unsigned int*>(Blds + (size_t)col * K);
    __builtin_prefetch(ap, 0, 1);

    v8f acc = {0.f, 0.f, 0.f, 0.f, 0.f, 0.f, 0.f, 0.f};
    for (int k0 = 0; k0 < K; k0 += 32) {
        FragU a, b;
        // A 16x32 bf16 frag: half i -> K = k0 + grp*16 + i
#pragma unroll
        for (int p = 0; p < 8; p++) {
            int i0 = 2 * p;
            int k = k0 + (grp << 4) + i0;
            a.u[p] = ap[k >> 1];
        }
        // B 32x16 bf16 frag: half i -> K = k0 + grp*16 + i, fixed col.
        // B^T row [col] is contiguous in K -> two b128 LDS reads.
        int kb = (k0 + (grp << 4)) >> 1;       // uint index
#pragma unroll
        for (int p = 0; p < 8; p++) b.u[p] = bp[kb + p];
        acc = __builtin_amdgcn_wmma_f32_16x16x32_bf16(
            false, a.v, false, b.v, (short)0, acc, false, false);
    }
    // D frag: VGPR i -> row = tm*16 + grp*8 + i, col = tn*16 + (lane&15)
#pragma unroll
    for (int i = 0; i < 8; i++) {
        int gm = (tm << 4) + (grp << 3) + i;
        float vv = acc[i];
        if (EPI == 0) {
            outH[(size_t)gm * N + col] = f2b(vv);
        } else if (EPI == 1) {
            size_t idx = (size_t)gm * N + col;
            outF[idx] = vv + bias[col] + res[idx];
        } else {
            float t = vv + bias[col];
            float g = 0.5f * t * (1.0f + erff(t * 0.70710678118f));
            outH[(size_t)gm * N + col] = f2b(g);
        }
    }
}

// ---------------------------------------------------------------------------
// Conv value stack.
// conv1: h[B,N,64] (bf16) * w[2,1,25] -> t1[B,2,N,40] (f32)
// ---------------------------------------------------------------------------
__global__ void conv1_kernel(const unsigned short* __restrict__ h,
                             const float* __restrict__ w,
                             const float* __restrict__ bias,
                             float* __restrict__ t1) {
    int idx = blockIdx.x * blockDim.x + threadIdx.x;
    const int total = Bn * 2 * Nn * 40;
    if (idx >= total) return;
    int d = idx % 40;
    int n = (idx / 40) % Nn;
    int c = (idx / (40 * Nn)) % 2;
    int b = idx / (40 * Nn * 2);
    float acc = bias[c];
    const unsigned short* hr = h + ((size_t)(b * Nn + n)) * DIMn + d;
#pragma unroll
    for (int j = 0; j < 25; j++) acc += b2f(hr[j]) * w[c * 25 + j];
    t1[idx] = acc;
}

// conv2: t1[B,2,N,40] * w[4,2,10] -> vT[B,128,512] bf16 (transposed value:
// vT[b][col][n], col = c*31+d, pad cols 124..127 = 0)
__global__ void conv2_kernel(const float* __restrict__ t1,
                             const float* __restrict__ w,
                             const float* __restrict__ bias,
                             unsigned short* __restrict__ vT) {
    int idx = blockIdx.x * blockDim.x + threadIdx.x;
    const int total = Bn * Nn * 128;
    if (idx >= total) return;
    int col = idx & 127;
    int n = (idx >> 7) % Nn;
    int b = (idx >> 7) / Nn;
    float r = 0.0f;
    if (col < CDIMn) {
        int c = col / 31, d = col % 31;
        float acc = bias[c];
#pragma unroll
        for (int cin = 0; cin < 2; cin++) {
            const float* tp = t1 + (((size_t)(b * 2 + cin) * Nn) + n) * 40 + d;
#pragma unroll
            for (int j = 0; j < 10; j++) acc += tp[j] * w[c * 20 + cin * 10 + j];
        }
        r = acc;
    }
    vT[((size_t)(b * 128 + col)) * Nn + n] = f2b(r);
}

// ---------------------------------------------------------------------------
// Fused attention: one block = one (b,h) and one 16-query tile.
//  phase1: S[16,512] = (Q @ K^T) * scale   via WMMA bf16 -> LDS (f32)
//  phase2: softmax per row (wave32 shuffles); P stored packed bf16 in LDS;
//          optionally emit attn (f32) to d_out
//  phase3: O[16,128] = P @ V via WMMA bf16; P A-frags are 2x ds_load_b128,
//          V B-frags contiguous b128 from transposed vT.
// ---------------------------------------------------------------------------
#define SLDS 520
__global__ void attn_kernel(const unsigned short* __restrict__ qk,
                            const unsigned short* __restrict__ vT,
                            unsigned short* __restrict__ o,
                            float* __restrict__ attn_out, int lastLayer) {
    __shared__ float S[16][SLDS];
    __shared__ unsigned short Pb[16][Nn];
    int bh = blockIdx.x;          // 0..127
    int mt = blockIdx.y;          // 0..31
    int b = bh >> 3, h = bh & 7;
    int wave = threadIdx.x >> 5, lane = threadIdx.x & 31;
    int lr = lane & 15, grp = lane >> 4;

    // ---- phase 1: S = Q K^T * 0.125 ------------------------------------
    FragU qa[2];
    {
        const unsigned int* qp = reinterpret_cast<const unsigned int*>(
            qk + ((size_t)(b * Nn + mt * 16 + lr)) * QKW + h * DHn);
#pragma unroll
        for (int kk = 0; kk < 2; kk++)
#pragma unroll
            for (int p = 0; p < 8; p++) {
                int i0 = 2 * p;
                int k = kk * 32 + (grp << 4) + i0;
                qa[kk].u[p] = qp[k >> 1];
            }
    }
#pragma unroll
    for (int j = 0; j < 4; j++) {
        int tn = wave + j * 8;                  // 0..31 key tiles
        int ncol = tn * 16 + lr;                // key row index (B-frag col)
        const unsigned int* kp = reinterpret_cast<const unsigned int*>(
            qk + ((size_t)(b * Nn + ncol)) * QKW + 512 + h * DHn);
        v8f acc = {0.f, 0.f, 0.f, 0.f, 0.f, 0.f, 0.f, 0.f};
#pragma unroll
        for (int kk = 0; kk < 2; kk++) {
            FragU kb;
#pragma unroll
            for (int p = 0; p < 8; p++) kb.u[p] = kp[kk * 16 + grp * 8 + p];
            acc = __builtin_amdgcn_wmma_f32_16x16x32_bf16(
                false, qa[kk].v, false, kb.v, (short)0, acc, false, false);
        }
#pragma unroll
        for (int i = 0; i < 8; i++) {
            int r = grp * 8 + i;
            S[r][tn * 16 + lr] = acc[i] * 0.125f;   // DH^-0.5
        }
    }
    __syncthreads();

    // ---- phase 2: softmax, each wave owns 2 rows; emit packed bf16 P ----
#pragma unroll
    for (int rr = 0; rr < 2; rr++) {
        int r = wave * 2 + rr;
        float vals[16];
        float mx = -1e30f;
#pragma unroll
        for (int t = 0; t < 16; t++) {
            vals[t] = S[r][t * 32 + lane];
            mx = fmaxf(mx, vals[t]);
        }
        for (int m = 16; m >= 1; m >>= 1) mx = fmaxf(mx, __shfl_xor(mx, m, 32));
        float sum = 0.f;
#pragma unroll
        for (int t = 0; t < 16; t++) { vals[t] = __expf(vals[t] - mx); sum += vals[t]; }
        for (int m = 16; m >= 1; m >>= 1) sum += __shfl_xor(sum, m, 32);
        float inv = 1.0f / sum;
#pragma unroll
        for (int t = 0; t < 16; t++) {
            float p = vals[t] * inv;
            Pb[r][t * 32 + lane] = f2b(p);
            if (lastLayer)
                attn_out[(((size_t)bh * Nn) + (mt * 16 + r)) * Nn + t * 32 + lane] = p;
        }
    }
    __syncthreads();

    // ---- phase 3: O = P @ V  (wave owns one 16-col tile of 128) ---------
    {
        int tn = wave;                           // 0..7
        int gn = tn * 16 + lr;
        const unsigned int* pp =
            reinterpret_cast<const unsigned int*>(&Pb[lr][0]);
        const unsigned int* vp = reinterpret_cast<const unsigned int*>(
            vT + ((size_t)(b * 128 + gn)) * Nn);
        v8f acc = {0.f, 0.f, 0.f, 0.f, 0.f, 0.f, 0.f, 0.f};
        for (int ks = 0; ks < 16; ks++) {        // K = 512 / 32
            FragU a, bb;
            // P A-frag: halves {ks*32+grp*16+0..15} -> 2x b128
            int base = (ks * 32 + grp * 16) >> 1; // uint index
#pragma unroll
            for (int p = 0; p < 4; p++) {
                a.u[p]     = pp[base + p];
                a.u[4 + p] = pp[base + 4 + p];
            }
            // V B-frag: kv = ks*32 + grp*16 + i contiguous in transposed vT
            int vb = (ks * 32 + (grp << 4)) >> 1;
#pragma unroll
            for (int p = 0; p < 8; p++) bb.u[p] = vp[vb + p];
            acc = __builtin_amdgcn_wmma_f32_16x16x32_bf16(
                false, a.v, false, bb.v, (short)0, acc, false, false);
        }
#pragma unroll
        for (int i = 0; i < 8; i++) {
            int r = grp * 8 + i;
            if (gn < CDIMn)
                o[((size_t)(b * Nn + mt * 16 + r)) * VDIMn + h * CDIMn + gn] =
                    f2b(acc[i]);
        }
    }
}

// ---------------------------------------------------------------------------
// Host side
// ---------------------------------------------------------------------------
extern "C" void kernel_launch(void* const* d_in, const int* in_sizes, int n_in,
                              void* d_out, int out_size, void* d_ws, size_t ws_size,
                              hipStream_t stream) {
    const float* x_in    = (const float*)d_in[0];
    const float* ln1_g   = (const float*)d_in[1];
    const float* ln1_b   = (const float*)d_in[2];
    const float* w_qk    = (const float*)d_in[3];
    const float* conv1_w = (const float*)d_in[4];
    const float* conv1_b = (const float*)d_in[5];
    const float* conv2_w = (const float*)d_in[6];
    const float* conv2_b = (const float*)d_in[7];
    const float* w_out   = (const float*)d_in[8];
    const float* b_out   = (const float*)d_in[9];
    const float* ln2_g   = (const float*)d_in[10];
    const float* ln2_b   = (const float*)d_in[11];
    const float* w1      = (const float*)d_in[12];
    const float* b1      = (const float*)d_in[13];
    const float* w2      = (const float*)d_in[14];
    const float* b2      = (const float*)d_in[15];

    float* xbuf = (float*)d_out;                     // running x [B,N,64]
    float* attn_out = (float*)d_out + (size_t)ROWS * DIMn;

    // workspace carve-up
    char* ws = (char*)d_ws;
    size_t off = 0;
    auto carve = [&](size_t bytes) {
        size_t o = off;
        off = (off + bytes + 255) & ~(size_t)255;
        return o;
    };
    unsigned short* h_bf  = (unsigned short*)(ws + carve((size_t)ROWS * DIMn * 2));
    unsigned short* qk_bf = (unsigned short*)(ws + carve((size_t)ROWS * QKW * 2));
    float*          t1    = (float*)        (ws + carve((size_t)Bn * 2 * Nn * 40 * 4));
    unsigned short* vT_bf = (unsigned short*)(ws + carve((size_t)Bn * 128 * Nn * 2));
    unsigned short* o_bf  = (unsigned short*)(ws + carve((size_t)ROWS * VDIMn * 2));
    unsigned short* f1_bf = (unsigned short*)(ws + carve((size_t)ROWS * FFn * 2));
    // transposed bf16 weights: [depth, N, K]
    unsigned short* wqkT  = (unsigned short*)(ws + carve((size_t)4 * QKW * DIMn * 2));
    unsigned short* woutT = (unsigned short*)(ws + carve((size_t)4 * DIMn * VDIMn * 2));
    unsigned short* w1T   = (unsigned short*)(ws + carve((size_t)4 * FFn * DIMn * 2));
    unsigned short* w2T   = (unsigned short*)(ws + carve((size_t)4 * DIMn * FFn * 2));

    // weight conversion f32 [depth,K,N] -> bf16 transposed [depth,N,K]
    auto cvtT = [&](const float* src, unsigned short* dst, int K, int N) {
        int total = 4 * K * N;
        cvt_transpose_kernel<<<(total + 255) / 256, 256, 0, stream>>>(
            src, dst, K, N, total);
    };
    cvtT(w_qk,  wqkT,  DIMn,  QKW);
    cvtT(w_out, woutT, VDIMn, DIMn);
    cvtT(w1,    w1T,   DIMn,  FFn);
    cvtT(w2,    w2T,   FFn,   DIMn);

    // x -> running buffer in d_out
    (void)hipMemcpyAsync(xbuf, x_in, (size_t)ROWS * DIMn * sizeof(float),
                         hipMemcpyDeviceToDevice, stream);

    auto gemm_blocks = [](int M, int N) {
        int tiles = (M >> 4) * (N >> 4);
        return (tiles + 7) / 8;                  // 8 waves per 256-thread block
    };

    for (int l = 0; l < 4; l++) {
        // 1) LN1 -> h (bf16)
        ln_kernel<<<ROWS / 8, 256, 0, stream>>>(xbuf, ln1_g + l * DIMn,
                                                ln1_b + l * DIMn, h_bf, ROWS);
        // 2) qk = h @ w_qk  -> bf16 [8192,1024]   (LDS-staged B^T: 128KB)
        gemm_wmma<0><<<gemm_blocks(ROWS, QKW), 256,
                       (size_t)QKW * DIMn * 2, stream>>>(
            h_bf, wqkT + (size_t)l * QKW * DIMn, ROWS, QKW, DIMn,
            nullptr, nullptr, nullptr, qk_bf);
        // 3) conv value stack (writes transposed vT)
        conv1_kernel<<<(Bn * 2 * Nn * 40 + 255) / 256, 256, 0, stream>>>(
            h_bf, conv1_w + l * 50, conv1_b + l * 2, t1);
        conv2_kernel<<<(Bn * Nn * 128 + 255) / 256, 256, 0, stream>>>(
            t1, conv2_w + l * 80, conv2_b + l * 4, vT_bf);
        // 4) fused attention (writes o, and attn on last layer)
        attn_kernel<<<dim3(Bn * HEADSn, Nn / 16), 256, 0, stream>>>(
            qk_bf, vT_bf, o_bf, attn_out, (l == 3) ? 1 : 0);
        // 5) x = o @ w_out + b_out + x   (f32, in place on xbuf; B^T 124KB)
        gemm_wmma<1><<<gemm_blocks(ROWS, DIMn), 256,
                       (size_t)DIMn * VDIMn * 2, stream>>>(
            o_bf, woutT + (size_t)l * DIMn * VDIMn, ROWS, DIMn, VDIMn,
            b_out + l * DIMn, xbuf, xbuf, nullptr);
        // 6) LN2 -> h (bf16, reuse buffer)
        ln_kernel<<<ROWS / 8, 256, 0, stream>>>(xbuf, ln2_g + l * DIMn,
                                                ln2_b + l * DIMn, h_bf, ROWS);
        // 7) ff1 = gelu(h @ w1 + b1) -> bf16   (B^T 32KB)
        gemm_wmma<2><<<gemm_blocks(ROWS, FFn), 256,
                       (size_t)FFn * DIMn * 2, stream>>>(
            h_bf, w1T + (size_t)l * FFn * DIMn, ROWS, FFn, DIMn,
            b1 + l * FFn, nullptr, nullptr, f1_bf);
        // 8) x = ff1 @ w2 + b2 + x   (B^T 32KB)
        gemm_wmma<1><<<gemm_blocks(ROWS, DIMn), 256,
                       (size_t)DIMn * FFn * 2, stream>>>(
            f1_bf, w2T + (size_t)l * DIMn * FFn, ROWS, DIMn, FFn,
            b2 + l * DIMn, xbuf, xbuf, nullptr);
    }
    (void)in_sizes; (void)n_in; (void)out_size; (void)ws_size;
}
